// FP8Matmul_7103875908081
// MI455X (gfx1250) — 8-bit matrix operand order repaired and checked by reading, compiled, not run on hardware
//
#include <hip/hip_runtime.h>
#include <hip/hip_bf16.h>
#include <stdint.h>


#define MDIM 16384
#define KDIM 2048
#define NDIM 2048
#define FP8MAX 448.0f

typedef __attribute__((ext_vector_type(16))) int      v16i;
typedef __attribute__((ext_vector_type(8)))  float    v8f;
typedef __attribute__((ext_vector_type(4)))  int      v4i;
typedef __attribute__((ext_vector_type(4)))  float    f4;
typedef __attribute__((ext_vector_type(4)))  uint32_t u4;
typedef __attribute__((address_space(1))) v4i gv4i;
typedef __attribute__((address_space(3))) v4i lv4i;

#if defined(__has_builtin)
#if __has_builtin(__builtin_amdgcn_global_load_async_to_lds_b128)
#define USE_ASYNC_LDS 1
#endif
#endif

__device__ __forceinline__ void wait_asynccnt0() {
#if defined(__has_builtin) && __has_builtin(__builtin_amdgcn_s_wait_asynccnt)
  __builtin_amdgcn_s_wait_asynccnt(0);
#else
  asm volatile("s_wait_asynccnt 0x0" ::: "memory");
#endif
}

#ifdef USE_ASYNC_LDS
__device__ __forceinline__ void async_cp16(const uint8_t* g, uint8_t* l) {
  __builtin_amdgcn_global_load_async_to_lds_b128((gv4i*)g, (lv4i*)l, 0, 0);
}
#endif

__device__ __forceinline__ uint32_t pack4_fp8(float a, float b, float c, float d) {
  a = fminf(fmaxf(a, -FP8MAX), FP8MAX);
  b = fminf(fmaxf(b, -FP8MAX), FP8MAX);
  c = fminf(fmaxf(c, -FP8MAX), FP8MAX);
  d = fminf(fmaxf(d, -FP8MAX), FP8MAX);
  int r = __builtin_amdgcn_cvt_pk_fp8_f32(a, b, 0, false);
  r     = __builtin_amdgcn_cvt_pk_fp8_f32(c, d, r, true);
  return (uint32_t)r;
}

__global__ __launch_bounds__(256) void quantA_kernel(
    const float* __restrict__ A, uint8_t* __restrict__ A8) {
  long base = ((long)blockIdx.x * 256 + threadIdx.x) * 16;
  const f4* src = (const f4*)(A + base);
  uint32_t r[4];
#pragma unroll
  for (int i = 0; i < 4; ++i) {
    f4 v = __builtin_nontemporal_load(src + i);
    r[i] = pack4_fp8(v.x, v.y, v.z, v.w);
  }
  u4 d = {r[0], r[1], r[2], r[3]};
  *(u4*)(A8 + base) = d;
}

__global__ __launch_bounds__(256) void quantBT_kernel(
    const float* __restrict__ B, uint8_t* __restrict__ BT) {
  __shared__ __align__(16) uint8_t tile[64 * 80];
  const int t  = threadIdx.x;
  const int n0 = (blockIdx.x & 31) * 64;
  const int k0 = (blockIdx.x >> 5) * 64;
#pragma unroll
  for (int i = 0; i < 4; ++i) {
    int idx = t + 256 * i;
    int kk  = idx >> 4;
    int nn  = (idx & 15) * 4;
    f4 v = __builtin_nontemporal_load(
        (const f4*)(B + (long)(k0 + kk) * NDIM + n0 + nn));
    uint32_t p = pack4_fp8(v.x, v.y, v.z, v.w);
    tile[(nn + 0) * 80 + kk] = (uint8_t)(p       & 0xff);
    tile[(nn + 1) * 80 + kk] = (uint8_t)((p >> 8)  & 0xff);
    tile[(nn + 2) * 80 + kk] = (uint8_t)((p >> 16) & 0xff);
    tile[(nn + 3) * 80 + kk] = (uint8_t)((p >> 24) & 0xff);
  }
  __syncthreads();
  int row = t >> 2;
  int c16 = (t & 3) * 16;
  u4 d = *(const u4*)(tile + row * 80 + c16);
  *(u4*)(BT + (long)(n0 + row) * KDIM + k0 + c16) = d;
}

#define LDS_PITCH 144
#define NKT (KDIM / 128)

__global__ __launch_bounds__(256) void gemm_fp8_kernel(
    const uint8_t* __restrict__ A8, const uint8_t* __restrict__ BT8,
    __hip_bfloat16* __restrict__ out) {
  __shared__ __align__(16) uint8_t ldsA[2][128 * LDS_PITCH];
  __shared__ __align__(16) uint8_t ldsB[2][128 * LDS_PITCH];

  const int t    = threadIdx.x;
  const int lane = t & 31;
  const int w    = t >> 5;
  const int bm   = (int)blockIdx.x >> 4;
  const int bn   = (int)blockIdx.x & 15;
  const long rowA0 = (long)bm * 128;
  const long rowB0 = (long)bn * 128;

  const int h     = lane >> 4;
  const int ml    = lane & 15;
  const int mwave = (w >> 1) * 32;
  const int nwave = (w & 1) * 64;

  const int r0   = t >> 3;
  const int col0 = (t & 7) * 16;

#ifndef USE_ASYNC_LDS
  u4 ra[4], rb[4];
#endif

#ifdef USE_ASYNC_LDS
#define STAGE_TILE(buf, kt)                                                     \
  {                                                                             \
    const long kb_ = (long)(kt) * 128;                                          \
    _Pragma("unroll") for (int i = 0; i < 4; ++i) {                             \
      int r = r0 + 32 * i;                                                      \
      async_cp16(A8  + (rowA0 + r) * KDIM + kb_ + col0,                         \
                 &ldsA[buf][r * LDS_PITCH + col0]);                             \
      async_cp16(BT8 + (rowB0 + r) * KDIM + kb_ + col0,                         \
                 &ldsB[buf][r * LDS_PITCH + col0]);                             \
    }                                                                           \
  }
#else
#define STAGE_LOAD(kt)                                                          \
  {                                                                             \
    const long kb_ = (long)(kt) * 128;                                          \
    _Pragma("unroll") for (int i = 0; i < 4; ++i) {                             \
      int r = r0 + 32 * i;                                                      \
      ra[i] = *(const u4*)(A8  + (rowA0 + r) * KDIM + kb_ + col0);              \
      rb[i] = *(const u4*)(BT8 + (rowB0 + r) * KDIM + kb_ + col0);              \
    }                                                                           \
  }
#define STAGE_STORE(buf)                                                        \
  {                                                                             \
    _Pragma("unroll") for (int i = 0; i < 4; ++i) {                             \
      int r = r0 + 32 * i;                                                      \
      *(u4*)(&ldsA[buf][r * LDS_PITCH + col0]) = ra[i];                         \
      *(u4*)(&ldsB[buf][r * LDS_PITCH + col0]) = rb[i];                         \
    }                                                                           \
  }
#endif

#ifdef USE_ASYNC_LDS
  STAGE_TILE(0, 0);
  wait_asynccnt0();
#else
  STAGE_LOAD(0);
  STAGE_STORE(0);
#endif
  __syncthreads();

  v8f acc[2][4] = {};

  for (int kt = 0; kt < NKT; ++kt) {
    const int buf = kt & 1;

#ifdef USE_ASYNC_LDS
    if (kt < NKT - 1) STAGE_TILE(buf ^ 1, kt + 1);
#else
    if (kt < NKT - 1) STAGE_LOAD(kt + 1);
#endif

    v16i afrag[2];
#pragma unroll
    for (int mt = 0; mt < 2; ++mt) {
      const uint8_t* base = &ldsA[buf][(mwave + mt * 16 + ml) * LDS_PITCH + 16 * h];
#pragma unroll
      for (int p = 0; p < 8; ++p) {
        uint2 d = *(const uint2*)(base + 32 * (p >> 1) + 8 * (p & 1));
        afrag[mt][2 * p]     = (int)d.x;
        afrag[mt][2 * p + 1] = (int)d.y;
      }
    }

#pragma unroll
    for (int nt = 0; nt < 4; ++nt) {
      v16i bfrag;
      const uint8_t* base = &ldsB[buf][(nwave + nt * 16 + ml) * LDS_PITCH + 16 * h];
#pragma unroll
      for (int q = 0; q < 4; ++q) {
        u4 d = *(const u4*)(base + 32 * q);
        bfrag[4 * q]     = (int)d.x;
        bfrag[4 * q + 1] = (int)d.y;
        bfrag[4 * q + 2] = (int)d.z;
        bfrag[4 * q + 3] = (int)d.w;
      }
#pragma unroll
      for (int mt = 0; mt < 2; ++mt)
        acc[mt][nt] = __builtin_amdgcn_wmma_f32_16x16x128_fp8_fp8(
            afrag[mt], bfrag, (short)0, acc[mt][nt], false, false);
    }

#ifdef USE_ASYNC_LDS
    if (kt < NKT - 1) wait_asynccnt0();
    __syncthreads();
#else
    __syncthreads();
    if (kt < NKT - 1) {
      STAGE_STORE(buf ^ 1);
      __syncthreads();
    }
#endif
  }

  uint8_t* ldsC = &ldsA[0][0];
#pragma unroll
  for (int mt = 0; mt < 2; ++mt) {
#pragma unroll
    for (int nt = 0; nt < 4; ++nt) {
      const int crow = mwave + mt * 16 + 8 * h;
      const int ccol = nwave + nt * 16 + ml;
#pragma unroll
      for (int r = 0; r < 8; ++r) {
        *(__hip_bfloat16*)(ldsC + (crow + r) * 256 + ccol * 2) =
            __float2bfloat16(acc[mt][nt][r]);
      }
    }
  }
  __syncthreads();
#pragma unroll
  for (int i = 0; i < 8; ++i) {
    int c   = t + 256 * i;
    int row = c >> 4;
    int off = (c & 15) * 16;
    u4 d = *(const u4*)(ldsC + row * 256 + off);
    u4* dst = (u4*)((uint8_t*)out + ((rowA0 + row) * NDIM + rowB0) * 2 + off);
    __builtin_nontemporal_store(d, dst);
  }
}

extern "C" void kernel_launch(void* const* d_in, const int* in_sizes, int n_in,
                              void* d_out, int out_size, void* d_ws, size_t ws_size,
                              hipStream_t stream) {
  const float* A = (const float*)d_in[0];
  const float* B = (const float*)d_in[1];
  __hip_bfloat16* out = (__hip_bfloat16*)d_out;

  uint8_t* A8  = (uint8_t*)d_ws;
  uint8_t* BT8 = A8 + (size_t)MDIM * KDIM;

  quantA_kernel<<<(MDIM * (long)KDIM) / (256 * 16), 256, 0, stream>>>(A, A8);
  quantBT_kernel<<<(KDIM / 64) * (NDIM / 64), 256, 0, stream>>>(B, BT8);
  gemm_fp8_kernel<<<(MDIM / 128) * (NDIM / 128), 256, 0, stream>>>(A8, BT8, out);
}
